// TwoMarginContrastiveLoss_27135603376234
// MI455X (gfx1250) — matrix operand order repaired and checked by reading, not compiled, not run on hardware
//
#include <hip/hip_runtime.h>
#include <hip/hip_bf16.h>

// ---------------------------------------------------------------------------
// TwoMarginContrastiveLoss for MI455X (gfx1250, wave32, WMMA)
//   sim = X X^T (8192x8192, D=256) fused with masked margin reduction.
//   Pass 0 converts X to f16 once (workspace); GEMM blocks stage f16 tiles
//   straight into LDS (no per-block conversion), then 16x16x32 f16 WMMA with
//   f32 accumulation. Deterministic two-stage reduction; fp32 last-row stats.
// ---------------------------------------------------------------------------

#define NN 8192
#define DD 256
#define MARGIN_SAME 0.9f
#define MARGIN_DIFF 0.5f

#define TILE 128
#define KPAD 8
#define PITCH (DD + KPAD)          // 264 halves per row -> 528B pitch (bank-spread)
#define BLOCKS_PER_DIM (NN / TILE) // 64
#define NUM_BLOCKS (BLOCKS_PER_DIM * BLOCKS_PER_DIM) // 4096

typedef __attribute__((ext_vector_type(16))) _Float16 v16h;
typedef __attribute__((ext_vector_type(8)))  _Float16 v8h;
typedef __attribute__((ext_vector_type(4)))  _Float16 v4h;
typedef __attribute__((ext_vector_type(8)))  float    v8f;

// ---------------------------------------------------------------------------
// Kernel 0: one-shot f32 -> f16 conversion of X into workspace (row major).
// 2M elements: 2048 blocks x 256 threads x 1 float4 each.
// ---------------------------------------------------------------------------
__global__ __launch_bounds__(256)
void convert_f16_kernel(const float* __restrict__ X, _Float16* __restrict__ Xh)
{
    const int idx = blockIdx.x * 256 + threadIdx.x;   // float4 slot, < 524288
    const float4 v = *(const float4*)(X + (size_t)idx * 4);
    v4h h = { (_Float16)v.x, (_Float16)v.y, (_Float16)v.z, (_Float16)v.w };
    *(v4h*)(Xh + (size_t)idx * 4) = h;
}

// ---------------------------------------------------------------------------
// Kernel 1: fused tiled GEMM (WMMA f16) + masked margin-loss partial sums
// grid = 4096 blocks of 256 threads (8 waves). Block tile: 128x128.
// F16SRC=true: stage pre-converted f16 (pure b128 copy through VGPRs).
// F16SRC=false: fallback, stage f32 with in-block conversion.
// ---------------------------------------------------------------------------
template <bool F16SRC>
__global__ __launch_bounds__(256)
void contrastive_gemm_loss_kernel(const float*    __restrict__ X,
                                  const _Float16* __restrict__ Xh,
                                  const int*      __restrict__ tgt,
                                  float*          __restrict__ partial)
{
    __shared__ __align__(16) _Float16 As[TILE * PITCH];
    __shared__ __align__(16) _Float16 Bs[TILE * PITCH];
    __shared__ float red[256];

    const int tid     = threadIdx.x;
    const int bcol    = blockIdx.x & (BLOCKS_PER_DIM - 1);
    const int brow    = blockIdx.x / BLOCKS_PER_DIM;
    const int rowBase = brow * TILE;
    const int colBase = bcol * TILE;

    if constexpr (F16SRC) {
        // 128x256 halves per tile = 4096 16B chunks; 16 per thread per tile.
        #pragma unroll 4
        for (int it = 0; it < 16; ++it) {
            const int idx = it * 256 + tid;   // 16B chunk in tile
            const int r   = idx >> 5;         // 32 chunks (of 8 halves) per row
            const int c   = idx & 31;
            const v8h ha = *(const v8h*)(Xh + (size_t)(rowBase + r) * DD + c * 8);
            const v8h hb = *(const v8h*)(Xh + (size_t)(colBase + r) * DD + c * 8);
            *(v8h*)(As + r * PITCH + c * 8) = ha;
            *(v8h*)(Bs + r * PITCH + c * 8) = hb;
        }
    } else {
        // 128*256/4 = 8192 float4 slots, 32 per thread; convert f32 -> f16.
        #pragma unroll 4
        for (int it = 0; it < 32; ++it) {
            const int idx = it * 256 + tid;
            const int r   = idx >> 6;         // 64 float4 per row
            const int c4  = idx & 63;
            const float4 va = *(const float4*)(X + (size_t)(rowBase + r) * DD + c4 * 4);
            const float4 vb = *(const float4*)(X + (size_t)(colBase + r) * DD + c4 * 4);
            v4h ha = { (_Float16)va.x, (_Float16)va.y, (_Float16)va.z, (_Float16)va.w };
            v4h hb = { (_Float16)vb.x, (_Float16)vb.y, (_Float16)vb.z, (_Float16)vb.w };
            *(v4h*)(As + r * PITCH + c4 * 4) = ha;
            *(v4h*)(Bs + r * PITCH + c4 * 4) = hb;
        }
    }
    __syncthreads();

    // -------- wave tiling: 8 waves -> 2 (M) x 4 (N); each wave: 64x32 of sim
    const int wid  = tid >> 5;
    const int lane = tid & 31;
    const int lh   = lane >> 4;    // lane half (K-group select per ISA layout)
    const int l16  = lane & 15;

    const int mBase = (wid & 1) * 64;   // 4 m-tiles of 16
    const int nBase = (wid >> 1) * 32;  // 2 n-tiles of 16

    v8f acc[4][2] = {};

    for (int kw = 0; kw < DD; kw += 32) {
        // B fragments: lane = column n; K = kw + lh*16 .. +15 contiguous (32B)
        v16h bfrag[2];
        #pragma unroll
        for (int tn = 0; tn < 2; ++tn) {
            const _Float16* bp = Bs + (nBase + tn * 16 + l16) * PITCH + kw + lh * 16;
            v16h b;
            ((v8h*)&b)[0] = *(const v8h*)(bp);
            ((v8h*)&b)[1] = *(const v8h*)(bp + 8);
            bfrag[tn] = b;
        }
        // A fragments: lane = row m; K = kw + lh*16 .. +15 contiguous (32B)
        v16h afrag[4];
        #pragma unroll
        for (int tm = 0; tm < 4; ++tm) {
            const _Float16* ap = As + (mBase + tm * 16 + l16) * PITCH + kw + lh * 16;
            v16h a;
            ((v8h*)&a)[0] = *(const v8h*)(ap);
            ((v8h*)&a)[1] = *(const v8h*)(ap + 8);
            afrag[tm] = a;
        }
        #pragma unroll
        for (int tm = 0; tm < 4; ++tm) {
            #pragma unroll
            for (int tn = 0; tn < 2; ++tn) {
                acc[tm][tn] = __builtin_amdgcn_wmma_f32_16x16x32_f16(
                    /*neg_a=*/false, afrag[tm],
                    /*neg_b=*/false, bfrag[tn],
                    /*c_mod=*/(short)0, acc[tm][tn],
                    /*reuse_a=*/false, /*reuse_b=*/false);
            }
        }
    }

    // -------- masked margin reduction over this wave's 64x32 sim tile.
    // C/D layout: lane L, VGPR r  ->  M = r + 8*(L>>4), N = L&15
    const int j0  = colBase + nBase + l16;
    const int j1  = j0 + 16;
    const int tj0 = tgt[j0];
    const int tj1 = tgt[j1];

    float local = 0.0f;
    #pragma unroll
    for (int tm = 0; tm < 4; ++tm) {
        #pragma unroll
        for (int r = 0; r < 8; ++r) {
            const int i  = rowBase + mBase + tm * 16 + r + lh * 8;
            const int ti = tgt[i];
            {
                const float s = acc[tm][0][r];
                if (ti == tj0) { if (i != j0 && s < MARGIN_SAME) local += 1.0f - s; }
                else           { if (s > MARGIN_DIFF)            local += s; }
            }
            {
                const float s = acc[tm][1][r];
                if (ti == tj1) { if (i != j1 && s < MARGIN_SAME) local += 1.0f - s; }
                else           { if (s > MARGIN_DIFF)            local += s; }
            }
        }
    }

    // -------- deterministic in-block tree reduction
    red[tid] = local;
    __syncthreads();
    #pragma unroll
    for (int off = 128; off > 0; off >>= 1) {
        if (tid < off) red[tid] += red[tid + off];
        __syncthreads();
    }
    if (tid == 0) partial[blockIdx.x] = red[0];
}

// ---------------------------------------------------------------------------
// Kernel 2: deterministic final reduction of 4096 partials -> loss, prec
// ---------------------------------------------------------------------------
__global__ __launch_bounds__(256)
void final_reduce_kernel(const float* __restrict__ partial, float* __restrict__ out)
{
    __shared__ float red[256];
    float s = 0.0f;
    for (int i = threadIdx.x; i < NUM_BLOCKS; i += 256) s += partial[i];
    red[threadIdx.x] = s;
    __syncthreads();
    #pragma unroll
    for (int off = 128; off > 0; off >>= 1) {
        if (threadIdx.x < off) red[threadIdx.x] += red[threadIdx.x + off];
        __syncthreads();
    }
    if (threadIdx.x == 0) {
        out[0] = red[0] * (1.0f / (float)NN);  // loss
        out[1] = 0.0f;                         // prec (never incremented in ref)
    }
}

// ---------------------------------------------------------------------------
// Kernel 3: last-row statistics in exact fp32 (matches reference semantics,
// including the sim < 1.0 diagonal quirk of pos_mask).
// ---------------------------------------------------------------------------
__global__ __launch_bounds__(256)
void last_row_stats_kernel(const float* __restrict__ X,
                           const int*   __restrict__ tgt,
                           float*       __restrict__ out)
{
    __shared__ __align__(16) float q[DD];
    __shared__ float rps[256], rpc[256], rns[256], rnc[256];

    q[threadIdx.x] = X[(size_t)(NN - 1) * DD + threadIdx.x];
    __syncthreads();

    const int tlast = tgt[NN - 1];
    float ps = 0.0f, pc = 0.0f, ns = 0.0f, nc = 0.0f;

    for (int j = threadIdx.x; j < NN; j += 256) {
        const float4* row4 = (const float4*)(X + (size_t)j * DD);
        const float4* q4   = (const float4*)q;
        float s = 0.0f;
        #pragma unroll 8
        for (int k = 0; k < DD / 4; ++k) {
            const float4 a = row4[k];
            const float4 b = q4[k];
            s += a.x * b.x + a.y * b.y + a.z * b.z + a.w * b.w;
        }
        if (tgt[j] == tlast) {
            if (s < 1.0f) { ps += s; pc += 1.0f; }   // pos_mask = same & sim<1
        } else {
            ns += s; nc += 1.0f;                     // neg_mask = ~same
        }
    }

    rps[threadIdx.x] = ps; rpc[threadIdx.x] = pc;
    rns[threadIdx.x] = ns; rnc[threadIdx.x] = nc;
    __syncthreads();
    #pragma unroll
    for (int off = 128; off > 0; off >>= 1) {
        if (threadIdx.x < off) {
            rps[threadIdx.x] += rps[threadIdx.x + off];
            rpc[threadIdx.x] += rpc[threadIdx.x + off];
            rns[threadIdx.x] += rns[threadIdx.x + off];
            rnc[threadIdx.x] += rnc[threadIdx.x + off];
        }
        __syncthreads();
    }
    if (threadIdx.x == 0) {
        out[2] = rps[0] / fmaxf(rpc[0], 1.0f);  // mean_pos_sim
        out[3] = rns[0] / fmaxf(rnc[0], 1.0f);  // mean_neg_sim
    }
}

// ---------------------------------------------------------------------------
extern "C" void kernel_launch(void* const* d_in, const int* in_sizes, int n_in,
                              void* d_out, int out_size, void* d_ws, size_t ws_size,
                              hipStream_t stream)
{
    const float* X   = (const float*)d_in[0];
    const int*   tgt = (const int*)d_in[1];
    float*       out = (float*)d_out;

    const size_t halfBytes = (size_t)NN * DD * sizeof(_Float16);      // 4 MB
    const size_t needBytes = halfBytes + (size_t)NUM_BLOCKS * sizeof(float);

    if (ws_size >= needBytes) {
        // fast path: one-shot f16 conversion, GEMM stages f16 directly
        _Float16* Xh      = (_Float16*)d_ws;
        float*    partial = (float*)((char*)d_ws + halfBytes);
        convert_f16_kernel<<<(NN * DD) / (256 * 4), 256, 0, stream>>>(X, Xh);
        contrastive_gemm_loss_kernel<true><<<NUM_BLOCKS, 256, 0, stream>>>(X, Xh, tgt, partial);
        final_reduce_kernel<<<1, 256, 0, stream>>>(partial, out);
    } else {
        // fallback: convert inside each block (no big workspace needed)
        float* partial = (float*)d_ws;
        contrastive_gemm_loss_kernel<false><<<NUM_BLOCKS, 256, 0, stream>>>(X, nullptr, tgt, partial);
        final_reduce_kernel<<<1, 256, 0, stream>>>(partial, out);
    }
    last_row_stats_kernel<<<1, 256, 0, stream>>>(X, tgt, out);
}
